// FeaturePyramid_74835510166129
// MI455X (gfx1250) — matrix operand order repaired and checked by reading, not compiled, not run on hardware
//
#include <hip/hip_runtime.h>

// ---------------------------------------------------------------------------
// CDNA5 (gfx1250, wave32) FeaturePyramid.
//   * every Cin%32==0 conv is implicit-GEMM on v_wmma_f32_16x16x32_bf16,
//     fully templated on shape so tap/chunk loops unroll and index math
//     constant-folds (many static v_wmma sites, no 64-bit div expansion)
//   * weights pre-packed into WMMA A-fragment lane order (2x b128/lane/chunk)
//   * activations kept as bf16 NHWC copies -> B fragment = 16 consecutive
//     bf16 channels (2x b128/lane/chunk), halving conv input traffic
//   * training-mode BN: LDS-atomic partial reduction (ds_add) + fused
//     normalize/residual/LeakyReLU writing fp32 + bf16 + optional NCHW
//   * correlation: in-place channel L2 normalize + 81-shift dot products
// ---------------------------------------------------------------------------

typedef __attribute__((ext_vector_type(16))) __bf16 v16bf;
typedef __attribute__((ext_vector_type(8)))  float  v8f;

#define LRELU_SLOPE 0.1f
#define BN_EPS 1e-5f

union F16x16 {            // 16 bf16 == 32 bytes == one WMMA fragment per lane
  uint4 u[2];
  v16bf v;
};

// ------------------------------ utility ------------------------------------

__global__ void k_zero(float* __restrict__ p, long n) {
  long i = (long)blockIdx.x * blockDim.x + threadIdx.x;
  if (i < n) p[i] = 0.f;
}

// --------------------- weight repack: OIHW fp32 -> WMMA bf16 ----------------
// dst[((tap*CH + ch)*Cout + m)*32 + h*16 + j] = w[m][ch*32 + K(j,h)][ky][kx]
// K(j,h) = 16h + j  (contiguous k order, matches B fragment)
__global__ void k_repack(const float* __restrict__ w, __bf16* __restrict__ dst,
                         int Cout, int Cin, int ks) {
  long idx = (long)blockIdx.x * blockDim.x + threadIdx.x;
  const long total = (long)Cout * Cin * ks * ks;
  if (idx >= total) return;
  const int i32 = (int)(idx & 31);
  const long r = idx >> 5;
  const int m = (int)(r % Cout);
  const long r2 = r / Cout;
  const int CH = Cin / 32;
  const int ch = (int)(r2 % CH);
  const int tap = (int)(r2 / CH);
  const int ky = tap / ks, kx = tap % ks;
  const int h = i32 >> 4, j = i32 & 15, v = j >> 1, e = j & 1;
  const int K = 16 * h + j;
  const int ci = ch * 32 + K;
  dst[idx] = (__bf16)w[(((long)m * Cin + ci) * ks + ky) * ks + kx];
}

// ------------------------- first conv (3->32, s2) --------------------------
__global__ void k_conv_first(const float* __restrict__ in,   // NCHW (B,3,Hin,Win)
                             const float* __restrict__ w,    // OIHW (32,3,3,3)
                             float* __restrict__ out,        // NHWC (B,Hout,Wout,32)
                             int Bn, int Hin, int Win, int Hout, int Wout) {
  long idx = (long)blockIdx.x * blockDim.x + threadIdx.x;
  long total = (long)Bn * Hout * Wout * 32;
  if (idx >= total) return;
  int co = (int)(idx & 31);
  long p = idx >> 5;
  int x = (int)(p % Wout);
  int y = (int)((p / Wout) % Hout);
  int b = (int)(p / ((long)Wout * Hout));
  float acc = 0.f;
  for (int ci = 0; ci < 3; ++ci)
    for (int ky = 0; ky < 3; ++ky)
      for (int kx = 0; kx < 3; ++kx) {
        int iy = 2 * y - 1 + ky, ix = 2 * x - 1 + kx;
        if (iy >= 0 && iy < Hin && ix >= 0 && ix < Win)
          acc += in[(((long)b * 3 + ci) * Hin + iy) * Win + ix] *
                 w[((co * 3 + ci) * 3 + ky) * 3 + kx];
      }
  out[idx] = acc;
}

// ------------------------ templated WMMA conv kernel -----------------------
// One wave per 16(Cout) x 16(pixel) tile. All shapes compile-time:
// divisions fold, tap/chunk loops unroll -> dense v_wmma + global_load_b128.
template <int Hin, int Win, int Cin, int Hout, int Wout, int Cout, int KS,
          int STRIDE, int PAD>
__global__ __launch_bounds__(128) void k_conv_wmma(
    const __bf16* __restrict__ in,   // bf16 NHWC (B,Hin,Win,Cin)
    const __bf16* __restrict__ wr,   // packed weights (see k_repack)
    const float* __restrict__ bias,  // nullptr or [Cout]
    float* __restrict__ out) {       // fp32 NHWC (B,Hout,Wout,Cout)
  constexpr int CH = Cin / 32;
  constexpr long HW = (long)Hout * Wout;
  const int wave = threadIdx.x >> 5;
  const int lane = threadIdx.x & 31;
  const int q = lane & 15;
  const int half = lane >> 4;
  const long pixTile = ((long)blockIdx.x * 4 + wave) * 16;
  const int coutBase = blockIdx.y * 16;

  const long np = pixTile + q;
  const int b = (int)(np / HW);           // constant divisor -> mul/shift
  const long rem = np % HW;
  const int oy = (int)(rem / Wout);
  const int ox = (int)(rem % Wout);
  const int iy0 = oy * STRIDE - PAD;
  const int ix0 = ox * STRIDE - PAD;

  // A rows for this lane: ((tap*CH+ch)*Cout + m)*32 + h*16
  const __bf16* wrow = wr + ((long)(coutBase + q)) * 32 + half * 16;
  constexpr long wstep = (long)Cout * 32;  // per (tap,chunk) advance

  v8f acc = {0.f, 0.f, 0.f, 0.f, 0.f, 0.f, 0.f, 0.f};
  const uint4 z4 = make_uint4(0u, 0u, 0u, 0u);

#pragma unroll
  for (int ky = 0; ky < KS; ++ky) {
#pragma unroll
    for (int kx = 0; kx < KS; ++kx) {
      const int iy = iy0 + ky, ix = ix0 + kx;
      const bool valid = (iy >= 0) & (iy < Hin) & (ix >= 0) & (ix < Win);
      const __bf16* ip =
          in + (((long)b * Hin + iy) * Win + ix) * Cin + 16 * half;
      const int tap = ky * KS + kx;
      const __bf16* wp = wrow + (long)tap * CH * wstep;
#pragma unroll
      for (int ch = 0; ch < CH; ++ch) {
        F16x16 a, bb;
        const uint4* ap = (const uint4*)(wp + (long)ch * wstep);
        a.u[0] = ap[0];
        a.u[1] = ap[1];
        if (valid) {
          const uint4* bp = (const uint4*)(ip + ch * 32);
          bb.u[0] = bp[0];
          bb.u[1] = bp[1];
        } else {
          bb.u[0] = z4;
          bb.u[1] = z4;
        }
        acc = __builtin_amdgcn_wmma_f32_16x16x32_bf16(
            /*neg_a=*/false, a.v, /*neg_b=*/false, bb.v,
            /*c_mod=*/(short)0, acc, /*reuse_a=*/false, /*reuse_b=*/false);
      }
    }
  }

  // D: VGPR r -> M = r + 8*half, N = q  => 8 contiguous channels per lane
  float* op = out + (((long)b * Hout + oy) * Wout + ox) * Cout + coutBase + 8 * half;
#pragma unroll
  for (int r = 0; r < 8; ++r) {
    float vv = acc[r];
    if (bias) vv += bias[coutBase + 8 * half + r];
    op[r] = vv;
  }
}

template <int Hin, int Win, int Cin, int Hout, int Wout, int Cout, int KS,
          int STRIDE, int PAD>
static void conv_launch(const __bf16* src, const __bf16* wr, const float* bias,
                        float* dst, hipStream_t stream) {
  dim3 grid((unsigned)((4L * Hout * Wout) / 64), (unsigned)(Cout / 16));
  k_conv_wmma<Hin, Win, Cin, Hout, Wout, Cout, KS, STRIDE, PAD>
      <<<grid, 128, 0, stream>>>(src, wr, bias, dst);
}

// ----------------------------- BatchNorm -----------------------------------

__global__ void k_bnstats_accum(const float* __restrict__ x,
                                float* __restrict__ accum, long total, int C) {
  __shared__ float s[512];
  for (int i = threadIdx.x; i < 2 * C; i += blockDim.x) s[i] = 0.f;
  __syncthreads();
  for (long e = (long)blockIdx.x * blockDim.x + threadIdx.x; e < total;
       e += (long)gridDim.x * blockDim.x) {
    const float v = x[e];
    const int c = (int)(e % C);
    atomicAdd(&s[c], v);  // ds_add on LDS
    atomicAdd(&s[C + c], v * v);
  }
  __syncthreads();
  for (int i = threadIdx.x; i < 2 * C; i += blockDim.x) atomicAdd(&accum[i], s[i]);
}

__global__ void k_bnstats_fin(const float* __restrict__ accum,
                              const float* __restrict__ g,
                              const float* __restrict__ bta,
                              float* __restrict__ scale,
                              float* __restrict__ shift, int C, long cnt) {
  const int c = threadIdx.x;
  if (c < C) {
    const float m = accum[c] / (float)cnt;
    const float var = accum[C + c] / (float)cnt - m * m;
    const float inv = rsqrtf(var + BN_EPS);
    const float sc = g[c] * inv;
    scale[c] = sc;
    shift[c] = bta[c] - m * sc;
  }
}

// fused: y = lrelu(raw*scale + shift [+ residual]); writes any of:
// fp32 NHWC master, bf16 NHWC copy (next conv input), fp32 NCHW (d_out).
__global__ void k_bn_act(const float* __restrict__ raw,
                         const float* __restrict__ sc,
                         const float* __restrict__ sh,
                         const float* __restrict__ res,
                         float* __restrict__ f32o,
                         __bf16* __restrict__ bfo,
                         float* __restrict__ nchw,
                         int Bn, int H, int W, int C) {
  long idx = (long)blockIdx.x * blockDim.x + threadIdx.x;
  const long total = (long)Bn * H * W * C;
  if (idx >= total) return;
  const int c = (int)(idx % C);
  const long p = idx / C;
  float v = raw[idx] * sc[c] + sh[c];
  if (res) v += res[idx];
  v = v >= 0.f ? v : LRELU_SLOPE * v;
  if (f32o) f32o[idx] = v;
  if (bfo) bfo[idx] = (__bf16)v;
  if (nchw) {
    const long hw = (long)H * W;
    nchw[((long)(p / hw) * C + c) * hw + (p % hw)] = v;
  }
}

// --------------------- bilinear upsample (2x) + add ------------------------
__global__ void k_upsample_add(float* __restrict__ dst,        // NHWC += up(src)
                               const float* __restrict__ src,  // NHWC
                               __bf16* __restrict__ bfo,       // bf16 copy of sum
                               int Bn, int Hd, int Wd, int C, int Hs, int Ws) {
  long idx = (long)blockIdx.x * blockDim.x + threadIdx.x;
  const long total = (long)Bn * Hd * Wd * C;
  if (idx >= total) return;
  const int c = (int)(idx % C);
  const long p = idx / C;
  const int x = (int)(p % Wd);
  const int y = (int)((p / Wd) % Hd);
  const int b = (int)(p / ((long)Wd * Hd));
  const float fy = (y + 0.5f) * (float)Hs / (float)Hd - 0.5f;
  const float fx = (x + 0.5f) * (float)Ws / (float)Wd - 0.5f;
  const int y0 = (int)floorf(fy), x0 = (int)floorf(fx);
  const float wy = fy - (float)y0, wx = fx - (float)x0;
  const int y0c = max(0, min(Hs - 1, y0)), y1c = max(0, min(Hs - 1, y0 + 1));
  const int x0c = max(0, min(Ws - 1, x0)), x1c = max(0, min(Ws - 1, x0 + 1));
  const float s00 = src[(((long)b * Hs + y0c) * Ws + x0c) * C + c];
  const float s01 = src[(((long)b * Hs + y0c) * Ws + x1c) * C + c];
  const float s10 = src[(((long)b * Hs + y1c) * Ws + x0c) * C + c];
  const float s11 = src[(((long)b * Hs + y1c) * Ws + x1c) * C + c];
  const float v = dst[idx] + (1.f - wy) * ((1.f - wx) * s00 + wx * s01) +
                  wy * ((1.f - wx) * s10 + wx * s11);
  dst[idx] = v;
  if (bfo) bfo[idx] = (__bf16)v;
}

__global__ void k_nhwc_to_nchw(const float* __restrict__ src, float* __restrict__ dst,
                               int Bn, int H, int W, int C) {
  long idx = (long)blockIdx.x * blockDim.x + threadIdx.x;
  const long total = (long)Bn * H * W * C;
  if (idx >= total) return;
  const int c = (int)(idx % C);
  const long p = idx / C;
  const long hw = (long)H * W;
  dst[((long)(p / hw) * C + c) * hw + (p % hw)] = src[idx];
}

// ------------------------------ correlation --------------------------------
__global__ void k_l2norm(float* __restrict__ buf, long P, int C) {
  long pix = (long)blockIdx.x * blockDim.x + threadIdx.x;
  if (pix >= P) return;
  float* p = buf + pix * (long)C;
  float s = 0.f;
  for (int c = 0; c < C; c += 4) {
    const float4 v = *(const float4*)(p + c);
    s += v.x * v.x + v.y * v.y + v.z * v.z + v.w * v.w;
  }
  const float inv = 1.f / fmaxf(sqrtf(s), 1e-12f);
  for (int c = 0; c < C; c += 4) {
    float4 v = *(float4*)(p + c);
    v.x *= inv; v.y *= inv; v.z *= inv; v.w *= inv;
    *(float4*)(p + c) = v;
  }
}

__global__ void k_corr(const float* __restrict__ n1, const float* __restrict__ n2,
                       float* __restrict__ out, int Bn, int H, int W, int C) {
  long idx = (long)blockIdx.x * blockDim.x + threadIdx.x;
  const long total = (long)Bn * 81 * H * W;
  if (idx >= total) return;
  const int x = (int)(idx % W);
  const int y = (int)((idx / W) % H);
  const int s = (int)((idx / ((long)W * H)) % 81);
  const int b = (int)(idx / ((long)81 * H * W));
  const int dy = s / 9 - 4, dx = s % 9 - 4;
  const int y2 = y + dy, x2 = x + dx;
  float acc = 0.f;
  if (y2 >= 0 && y2 < H && x2 >= 0 && x2 < W) {
    const float* a = n1 + (((long)b * H + y) * W + x) * C;
    const float* c2 = n2 + (((long)b * H + y2) * W + x2) * C;
    for (int c = 0; c < C; c += 4) {
      const float4 u = *(const float4*)(a + c);
      const float4 v = *(const float4*)(c2 + c);
      acc += u.x * v.x + u.y * v.y + u.z * v.z + u.w * v.w;
    }
  }
  out[idx] = acc;
}

// ------------------------------ orchestration ------------------------------
extern "C" void kernel_launch(void* const* d_in, const int* in_sizes, int n_in,
                              void* d_out, int out_size, void* d_ws, size_t ws_size,
                              hipStream_t stream) {
  (void)in_sizes; (void)n_in; (void)out_size; (void)ws_size;
  const int B = 4;
  const long HW0 = 128L * 224, HW1 = 64L * 112, HW2 = 32L * 56, HW3 = 16L * 28;
  auto in = [&](int i) { return (const float*)d_in[i]; };
  float* out = (float*)d_out;

  float* wsf = (float*)d_ws;
  size_t cur = 0;
  auto alloc = [&](long n) {
    float* p = wsf + cur;
    cur += ((size_t)n + 63) & ~(size_t)63;
    return p;
  };
  auto allocbf = [&](long n) {  // n bf16 elements
    __bf16* p = (__bf16*)(wsf + cur);
    cur += ((size_t)((n + 1) / 2) + 63) & ~(size_t)63;
    return p;
  };

  float* mp0[2]; float* mp1[2]; float* mp2[2]; float* mp3[2];
  for (int f = 0; f < 2; ++f) {
    mp0[f] = alloc(B * HW0 * 128); mp1[f] = alloc(B * HW1 * 128);
    mp2[f] = alloc(B * HW2 * 128); mp3[f] = alloc(B * HW3 * 128);
  }
  float*  RAW   = alloc(B * HW0 * 128);
  float*  LBUF  = alloc(B * HW0 * 128);
  float*  Ab    = alloc(B * HW0 * 32);
  float*  Ub    = alloc(B * HW3 * 256);
  __bf16* Abb   = allocbf(B * HW0 * 32);
  __bf16* Tbb   = allocbf(B * HW0 * 32);
  __bf16* Ubb   = allocbf(B * HW3 * 256);
  __bf16* F0b   = allocbf(B * HW0 * 32);
  __bf16* F1b   = allocbf(B * HW1 * 64);
  __bf16* F2b   = allocbf(B * HW2 * 128);
  __bf16* F3b   = allocbf(B * HW3 * 256);
  __bf16* LBUFb = allocbf(B * HW0 * 128);
  float*  stA   = alloc(512);
  float*  stS   = alloc(256);
  float*  stSh  = alloc(256);

  struct LayerCfg { int wi, Cout, Cin, ks; };
  const LayerCfg lay[] = {
      {5, 32, 32, 3},  {8, 32, 32, 3},  {11, 64, 32, 3},  {14, 64, 64, 3},
      {17, 64, 64, 3}, {20, 128, 64, 3}, {23, 128, 128, 3}, {26, 128, 128, 3},
      {29, 256, 128, 3}, {32, 256, 256, 3}, {35, 256, 256, 3}, {38, 256, 256, 3},
      {41, 256, 256, 3}, {44, 128, 256, 1}, {46, 128, 128, 1}, {48, 128, 64, 1},
      {50, 128, 32, 1}, {52, 128, 128, 3}, {55, 128, 128, 3}, {58, 128, 128, 3}};
  __bf16* wrp[61] = {};
  for (const auto& L : lay) {
    const long n = (long)L.Cout * L.Cin * L.ks * L.ks;
    wrp[L.wi] = allocbf(n);
    k_repack<<<dim3((unsigned)((n + 255) / 256)), 256, 0, stream>>>(
        in(L.wi), wrp[L.wi], L.Cout, L.Cin, L.ks);
  }

  auto zero = [&](float* p, long n) {
    k_zero<<<dim3((unsigned)((n + 255) / 256)), 256, 0, stream>>>(p, n);
  };
  auto bnfuse = [&](const float* raw, int gi, int bi, const float* res,
                    float* f32o, __bf16* bfo, float* nchw, int H, int W, int C) {
    zero(stA, 2L * C);
    const long tot = (long)B * H * W * C;
    k_bnstats_accum<<<dim3(768), 256, 0, stream>>>(raw, stA, tot, C);
    k_bnstats_fin<<<dim3(1), 256, 0, stream>>>(stA, in(gi), in(bi), stS, stSh, C,
                                               (long)B * H * W);
    k_bn_act<<<dim3((unsigned)((tot + 255) / 256)), 256, 0, stream>>>(
        raw, stS, stSh, res, f32o, bfo, nchw, B, H, W, C);
  };
  auto up_add = [&](float* dst, const float* src, __bf16* bfo, int Hd, int Wd,
                    int C, int Hs, int Ws) {
    const long n = (long)B * Hd * Wd * C;
    k_upsample_add<<<dim3((unsigned)((n + 255) / 256)), 256, 0, stream>>>(
        dst, src, bfo, B, Hd, Wd, C, Hs, Ws);
  };

  auto run_frame = [&](const float* frame, float* p0, float* p1, float* p2,
                       float* p3, float* featOut) {
    // ---- encoder level 0 ----
    { const long n = (long)B * HW0 * 32;
      k_conv_first<<<dim3((unsigned)((n + 255) / 256)), 256, 0, stream>>>(
          frame, in(2), RAW, B, 256, 448, 128, 224); }
    bnfuse(RAW, 3, 4, nullptr, Ab, Abb, nullptr, 128, 224, 32);
    conv_launch<128,224,32,128,224,32,3,1,1>(Abb, wrp[5], nullptr, RAW, stream);
    bnfuse(RAW, 6, 7, nullptr, nullptr, Tbb, nullptr, 128, 224, 32);
    conv_launch<128,224,32,128,224,32,3,1,1>(Tbb, wrp[8], nullptr, RAW, stream);
    bnfuse(RAW, 9, 10, Ab, nullptr, F0b, nullptr, 128, 224, 32);
    // ---- level 1 ----
    conv_launch<128,224,32,64,112,64,3,2,1>(F0b, wrp[11], nullptr, RAW, stream);
    bnfuse(RAW, 12, 13, nullptr, Ab, Abb, nullptr, 64, 112, 64);
    conv_launch<64,112,64,64,112,64,3,1,1>(Abb, wrp[14], nullptr, RAW, stream);
    bnfuse(RAW, 15, 16, nullptr, nullptr, Tbb, nullptr, 64, 112, 64);
    conv_launch<64,112,64,64,112,64,3,1,1>(Tbb, wrp[17], nullptr, RAW, stream);
    bnfuse(RAW, 18, 19, Ab, nullptr, F1b, nullptr, 64, 112, 64);
    // ---- level 2 ----
    conv_launch<64,112,64,32,56,128,3,2,1>(F1b, wrp[20], nullptr, RAW, stream);
    bnfuse(RAW, 21, 22, nullptr, Ab, Abb, nullptr, 32, 56, 128);
    conv_launch<32,56,128,32,56,128,3,1,1>(Abb, wrp[23], nullptr, RAW, stream);
    bnfuse(RAW, 24, 25, nullptr, nullptr, Tbb, nullptr, 32, 56, 128);
    conv_launch<32,56,128,32,56,128,3,1,1>(Tbb, wrp[26], nullptr, RAW, stream);
    bnfuse(RAW, 27, 28, Ab, nullptr, F2b, nullptr, 32, 56, 128);
    // ---- level 3 (two res blocks) ----
    conv_launch<32,56,128,16,28,256,3,2,1>(F2b, wrp[29], nullptr, RAW, stream);
    bnfuse(RAW, 30, 31, nullptr, Ab, Abb, nullptr, 16, 28, 256);
    conv_launch<16,28,256,16,28,256,3,1,1>(Abb, wrp[32], nullptr, RAW, stream);
    bnfuse(RAW, 33, 34, nullptr, nullptr, Tbb, nullptr, 16, 28, 256);
    conv_launch<16,28,256,16,28,256,3,1,1>(Tbb, wrp[35], nullptr, RAW, stream);
    bnfuse(RAW, 36, 37, Ab, Ub, Ubb, nullptr, 16, 28, 256);
    conv_launch<16,28,256,16,28,256,3,1,1>(Ubb, wrp[38], nullptr, RAW, stream);
    bnfuse(RAW, 39, 40, nullptr, nullptr, Tbb, nullptr, 16, 28, 256);
    conv_launch<16,28,256,16,28,256,3,1,1>(Tbb, wrp[41], nullptr, RAW, stream);
    bnfuse(RAW, 42, 43, Ub, nullptr, F3b, nullptr, 16, 28, 256);
    // ---- FPN ----
    conv_launch<16,28,256,16,28,128,1,1,0>(F3b, wrp[44], in(45), p3, stream);
    { const long n = (long)B * HW3 * 128;
      k_nhwc_to_nchw<<<dim3((unsigned)((n + 255) / 256)), 256, 0, stream>>>(
          p3, featOut + 19267584L, B, 16, 28, 128); }
    // p2
    conv_launch<32,56,128,32,56,128,1,1,0>(F2b, wrp[46], in(47), LBUF, stream);
    up_add(LBUF, p3, LBUFb, 32, 56, 128, 16, 28);
    conv_launch<32,56,128,32,56,128,3,1,1>(LBUFb, wrp[52], nullptr, RAW, stream);
    bnfuse(RAW, 53, 54, nullptr, p2, nullptr, featOut + 18350080L, 32, 56, 128);
    // p1
    conv_launch<64,112,64,64,112,128,1,1,0>(F1b, wrp[48], in(49), LBUF, stream);
    up_add(LBUF, p2, LBUFb, 64, 112, 128, 32, 56);
    conv_launch<64,112,128,64,112,128,3,1,1>(LBUFb, wrp[55], nullptr, RAW, stream);
    bnfuse(RAW, 56, 57, nullptr, p1, nullptr, featOut + 14680064L, 64, 112, 128);
    // p0
    conv_launch<128,224,32,128,224,128,1,1,0>(F0b, wrp[50], in(51), LBUF, stream);
    up_add(LBUF, p1, LBUFb, 128, 224, 128, 64, 112);
    conv_launch<128,224,128,128,224,128,3,1,1>(LBUFb, wrp[58], nullptr, RAW, stream);
    bnfuse(RAW, 59, 60, nullptr, p0, nullptr, featOut + 0L, 128, 224, 128);
  };

  run_frame(in(0), mp0[0], mp1[0], mp2[0], mp3[0], out);
  run_frame(in(1), mp0[1], mp1[1], mp2[1], mp3[1], out + 19496960L);

  // ---- correlation volumes ----
  auto corr = [&](float* a, float* b, float* o, int H, int W) {
    const long P = (long)B * H * W;
    k_l2norm<<<dim3((unsigned)((P + 255) / 256)), 256, 0, stream>>>(a, P, 128);
    k_l2norm<<<dim3((unsigned)((P + 255) / 256)), 256, 0, stream>>>(b, P, 128);
    const long n = (long)B * 81 * H * W;
    k_corr<<<dim3((unsigned)((n + 255) / 256)), 256, 0, stream>>>(a, b, o, B, H, W, 128);
  };
  corr(mp0[0], mp0[1], out + 38993920L, 128, 224);
  corr(mp1[0], mp1[1], out + 48283648L, 64, 112);
  corr(mp2[0], mp2[1], out + 50606080L, 32, 56);
  corr(mp3[0], mp3[1], out + 51186688L, 16, 28);
}
